// TriVAE_D3T_38826504356009
// MI455X (gfx1250) — matrix operand order repaired and checked by reading, not compiled, not run on hardware
//
#include <hip/hip_runtime.h>
#include <math.h>
#include <stdint.h>

// ---------------------------------------------------------------------------
// TriVAE forward for MI455X (gfx1250).
//  - All GEMMs: v_wmma_f32_16x16x32_f16, double-buffered LDS, fused epilogues.
//  - Attention: q staged via global_load_async_to_lds_b32 (ASYNCcnt),
//               k/v staged via the Tensor Data Mover (TENSORcnt).
// ---------------------------------------------------------------------------

typedef __attribute__((ext_vector_type(16))) _Float16 v16h;
typedef __attribute__((ext_vector_type(4)))  _Float16 h4;
typedef __attribute__((ext_vector_type(8)))  float    v8f;
typedef __attribute__((ext_vector_type(4)))  unsigned int v4u;
typedef __attribute__((ext_vector_type(8)))  int v8i_;
typedef __attribute__((ext_vector_type(4)))  int v4i_;

#define kB     2
#define kCIN   4
#define kEMB   256
#define kHh    120
#define kWw    120
#define kDd    64
#define kHP    30
#define kWP    30
#define kDP    16
#define kTOK   (kB * kHP * kWP * kDP)   // 28800 tokens
#define kNWIN  (kB * 6 * 6 * 4)         // 288 windows
#define kWN    100                      // tokens per 5x5x4 window
#define kHEADS 8
#define kDH    32

static __device__ __forceinline__ float gelu_f(float x) {
  return 0.5f * x * (1.0f + erff(x * 0.70710678118654752f));
}

// ---------------------------------------------------------------------------
// TDM 2D tile load: rows x rowlen f32 tile, row stride `stride` f32,
// global -> LDS.  D# layout per CDNA5 ISA.
// Issue from a single wave; completion via s_wait_tensorcnt.
// NOTE: this toolchain exposes the 6-arg builtin (g0,g1,g2,g3,g4,cpol).
// ---------------------------------------------------------------------------
static __device__ __forceinline__ void tdm_load_2d(uint32_t lds, const float* gptr,
                                                   int rows, int rowlen, int stride) {
  uint64_t ga = (uint64_t)(uintptr_t)gptr;
  v4u g0;
  g0[0] = 1u;                                   // count=1, user descriptor
  g0[1] = lds;                                  // lds_addr (bytes)
  g0[2] = (uint32_t)ga;                         // global_addr[31:0]
  g0[3] = (uint32_t)((ga >> 32) & 0x01FFFFFFu)  // global_addr[56:32]
          | (2u << 30);                         // type = 2 ("image")
  v8i_ g1;
  g1[0] = (2 << 16);                            // data_size = 4B; no multicast
  g1[1] = (rowlen & 0xFFFF) << 16;              // tensor_dim0[15:0]
  g1[2] = ((rowlen >> 16) & 0xFFFF)             // tensor_dim0[31:16]
          | ((rows & 0xFFFF) << 16);            // tensor_dim1[15:0]
  g1[3] = ((rows >> 16) & 0xFFFF)               // tensor_dim1[31:16]
          | ((rowlen & 0xFFFF) << 16);          // tile_dim0
  g1[4] = rows & 0xFFFF;                        // tile_dim1 (tile_dim2 = 0)
  g1[5] = stride;                               // tensor_dim0_stride[31:0]
  g1[6] = 0;                                    // strides hi = 0
  g1[7] = 0;
  v4i_ zz = {0, 0, 0, 0};
  v8i_ z8 = {0, 0, 0, 0, 0, 0, 0, 0};
  __builtin_amdgcn_tensor_load_to_lds(g0, g1, zz, zz, z8, 0);
}

// ---------------------------------------------------------------------------
// GEMM tile staging (f32 global -> f16 LDS), fast vectorized path for
// interior blocks: batched b128 loads, then cvt + b64 LDS stores.
// ---------------------------------------------------------------------------
static __device__ __forceinline__ void gemm_stage(
    const float* __restrict__ A, const float* __restrict__ Wt,
    _Float16 (* __restrict__ sA)[36], _Float16 (* __restrict__ sB)[36],
    int m0, int n0, int M, int N, int K, int k0, int wkn,
    bool fastA, bool fastB, int tid) {
  if (fastA) {
    float4 va[4];
#pragma unroll
    for (int t = 0; t < 4; ++t) {
      int idx = tid + t * 256;           // 0..1023 float4 slots of 128x32
      int r = idx >> 3, c4 = idx & 7;
      const float* ap = A + (size_t)(m0 + r) * K + k0 + c4 * 4;
      va[t] = *(const float4*)ap;
      __builtin_prefetch(ap + 32, 0, 1);  // next K-tile (speculative, DEV scope)
    }
#pragma unroll
    for (int t = 0; t < 4; ++t) {
      int idx = tid + t * 256;
      int r = idx >> 3, c4 = idx & 7;
      h4 hv;
      hv[0] = (_Float16)va[t].x; hv[1] = (_Float16)va[t].y;
      hv[2] = (_Float16)va[t].z; hv[3] = (_Float16)va[t].w;
      *(h4*)(&sA[r][c4 * 4]) = hv;
    }
  } else {
#pragma unroll 4
    for (int i = tid; i < 128 * 32; i += 256) {
      int r = i >> 5, c = i & 31;
      int gm = m0 + r;
      float v = (gm < M) ? A[(size_t)gm * K + (k0 + c)] : 0.0f;
      sA[r][c] = (_Float16)v;
    }
  }
  if (fastB) {
    float4 vb[2];
#pragma unroll
    for (int t = 0; t < 2; ++t) {
      int idx = tid + t * 256;           // 0..511 float4 slots of 64x32
      int r = idx >> 3, c4 = idx & 7;
      vb[t] = *(const float4*)(Wt + (size_t)(n0 + r) * K + k0 + c4 * 4);
    }
#pragma unroll
    for (int t = 0; t < 2; ++t) {
      int idx = tid + t * 256;
      int r = idx >> 3, c4 = idx & 7;
      h4 hv;
      hv[0] = (_Float16)vb[t].x; hv[1] = (_Float16)vb[t].y;
      hv[2] = (_Float16)vb[t].z; hv[3] = (_Float16)vb[t].w;
      *(h4*)(&sB[r][c4 * 4]) = hv;
    }
  } else if (wkn) {
    // W[K,N]: consecutive lanes read consecutive n (coalesced), scatter to LDS
#pragma unroll 4
    for (int i = tid; i < 64 * 32; i += 256) {
      int k = i >> 6, n = i & 63;
      int gn = n0 + n;
      float v = (gn < N) ? Wt[(size_t)(k0 + k) * N + gn] : 0.0f;
      sB[n][k] = (_Float16)v;
    }
  } else {
#pragma unroll 4
    for (int i = tid; i < 64 * 32; i += 256) {
      int r = i >> 5, c = i & 31;
      int gn = n0 + r;
      float v = (gn < N) ? Wt[(size_t)gn * K + (k0 + c)] : 0.0f;
      sB[r][c] = (_Float16)v;
    }
  }
}

// ---------------------------------------------------------------------------
// Generic WMMA GEMM: C[M,N] = act(A[M,K] @ W^T + bias) (+ res)
//   W layout: wkn==0 -> W[N,K] row-major (nn.Linear);  wkn==1 -> W[K,N].
// 256 threads = 8 wave32 waves; tile 128(M) x 64(N); K step 32;
// double-buffered LDS, one barrier per K-step.
// ---------------------------------------------------------------------------
__global__ __launch_bounds__(256) void k_gemm(
    const float* __restrict__ A, const float* __restrict__ Wt,
    const float* __restrict__ bias, const float* __restrict__ res,
    float* __restrict__ C, int M, int N, int K, int wkn, int act) {
  __shared__ _Float16 sA[2][128][36];
  __shared__ _Float16 sB[2][64][36];
  const int tid  = threadIdx.x;
  const int m0   = blockIdx.y * 128;
  const int n0   = blockIdx.x * 64;
  const int wid  = tid >> 5, lane = tid & 31;
  const int wm   = (wid & 3) * 32;
  const int wn   = (wid >> 2) * 32;

  const bool fastA = (m0 + 128 <= M);
  const bool fastB = (wkn == 0) && (n0 + 64 <= N);

  v8f acc[2][2] = {};

  gemm_stage(A, Wt, sA[0], sB[0], m0, n0, M, N, K, 0, wkn, fastA, fastB, tid);
  __syncthreads();

  const int nk = K >> 5;
  for (int kk = 0; kk < nk; ++kk) {
    const int cur = kk & 1;
    if (kk + 1 < nk)
      gemm_stage(A, Wt, sA[cur ^ 1], sB[cur ^ 1], m0, n0, M, N, K, (kk + 1) << 5,
                 wkn, fastA, fastB, tid);

    // A fragment: lanes 0-15 -> M=lane, K in {kb..kb+7, kb+16..kb+23}, kb=8*(lane>=16)
    const int arow = lane & 15, akb = (lane >> 4) << 3;
    v16h af[2];
#pragma unroll
    for (int s = 0; s < 2; ++s) {
      int m = wm + s * 16 + arow;
#pragma unroll
      for (int t = 0; t < 8; ++t) {
        af[s][t]     = sA[cur][m][akb + t];
        af[s][8 + t] = sA[cur][m][akb + 16 + t];
      }
    }
    // B fragment: lane -> N=lane&15, K in {kb..kb+7, kb+16..kb+23}, kb=8*(lane>=16)
    const int bcol = lane & 15, bkb = (lane >> 4) << 3;
    v16h bf[2];
#pragma unroll
    for (int s = 0; s < 2; ++s) {
      int n = wn + s * 16 + bcol;
#pragma unroll
      for (int t = 0; t < 8; ++t) {
        bf[s][t]     = sB[cur][n][bkb + t];
        bf[s][8 + t] = sB[cur][n][bkb + 16 + t];
      }
    }
#pragma unroll
    for (int i = 0; i < 2; ++i)
#pragma unroll
      for (int j = 0; j < 2; ++j)
        acc[i][j] = __builtin_amdgcn_wmma_f32_16x16x32_f16(
            false, af[i], false, bf[j], (short)0, acc[i][j], false, false);
    __syncthreads();
  }

  // C layout: N = lane&15; M = 8*(lane>=16) + r
  const int cn = lane & 15, cm = (lane >> 4) << 3;
#pragma unroll
  for (int i = 0; i < 2; ++i)
#pragma unroll
    for (int j = 0; j < 2; ++j)
#pragma unroll
      for (int r = 0; r < 8; ++r) {
        int m = m0 + wm + i * 16 + cm + r;
        int n = n0 + wn + j * 16 + cn;
        if (m < M && n < N) {
          float v = acc[i][j][r];
          if (bias) v += bias[n];
          if (act)  v = gelu_f(v);
          if (res)  v += res[(size_t)m * N + n];
          C[(size_t)m * N + n] = v;
        }
      }
}

// ---------------------------------------------------------------------------
// LayerNorm over C=256; one wave32 per row, 8 elems/lane, shfl reductions.
// ---------------------------------------------------------------------------
__global__ __launch_bounds__(256) void k_layernorm(
    const float* __restrict__ x, const float* __restrict__ g,
    const float* __restrict__ b, float* __restrict__ y, int rows) {
  int wid = threadIdx.x >> 5, lane = threadIdx.x & 31;
  int row = blockIdx.x * 8 + wid;
  if (row >= rows) return;
  const float* xr = x + (size_t)row * kEMB;
  float v[8];
  float s = 0.0f;
#pragma unroll
  for (int t = 0; t < 8; ++t) { v[t] = xr[lane + t * 32]; s += v[t]; }
#pragma unroll
  for (int o = 16; o > 0; o >>= 1) s += __shfl_xor(s, o);
  float mu = s * (1.0f / 256.0f);
  float var = 0.0f;
#pragma unroll
  for (int t = 0; t < 8; ++t) { float d = v[t] - mu; var += d * d; }
#pragma unroll
  for (int o = 16; o > 0; o >>= 1) var += __shfl_xor(var, o);
  var *= (1.0f / 256.0f);
  float inv = rsqrtf(var + 1e-5f);
  float* yr = y + (size_t)row * kEMB;
#pragma unroll
  for (int t = 0; t < 8; ++t) {
    int c = lane + t * 32;
    yr[c] = (v[t] - mu) * inv * g[c] + b[c];
  }
}

// ---------------------------------------------------------------------------
// Window partition with optional cyclic shift (roll by -SHIFT folded in).
// ---------------------------------------------------------------------------
__global__ void k_winpart(const float* __restrict__ src, float* __restrict__ dst,
                          int shifted) {
  int gid = blockIdx.x * 256 + threadIdx.x;  // kTOK*64 total
  if (gid >= kTOK * 64) return;
  int row = gid >> 6, c4 = gid & 63;
  int b = row / 14400, loc = row % 14400;
  int wl = loc / 100, t = loc % 100;
  int wh = wl / 24, ww = (wl >> 2) % 6, wd = wl & 3;
  int i0 = t / 20, i1 = (t >> 2) % 5, i2 = t & 3;
  int g0 = wh * 5 + i0, g1 = ww * 5 + i1, g2 = wd * 4 + i2;
  if (shifted) { g0 = (g0 + 2) % 30; g1 = (g1 + 2) % 30; g2 = (g2 + 2) % 16; }
  size_t s = (((size_t)((b * 30 + g0) * 30 + g1) * 16 + g2)) * 64 + c4;
  ((float4*)dst)[gid] = ((const float4*)src)[s];
}

// window reverse + inverse roll + residual add: out = shortcut + reverse(y)
__global__ void k_winrev(const float* __restrict__ shortcut,
                         const float* __restrict__ y, float* __restrict__ out,
                         int shifted) {
  int gid = blockIdx.x * 256 + threadIdx.x;  // kTOK*64 total
  if (gid >= kTOK * 64) return;
  int row = gid >> 6, c4 = gid & 63;
  int b = row / 14400, loc = row % 14400;
  int h = loc / 480, w = (loc >> 4) % 30, d = loc & 15;
  int g0 = h, g1 = w, g2 = d;
  if (shifted) { g0 = (h + 28) % 30; g1 = (w + 28) % 30; g2 = (d + 14) % 16; }
  int wh = g0 / 5, i0 = g0 % 5, ww = g1 / 5, i1 = g1 % 5, wd = g2 >> 2, i2 = g2 & 3;
  size_t wrow = (size_t)(b * 144 + (wh * 6 + ww) * 4 + wd) * 100 +
                (i0 * 5 + i1) * 4 + i2;
  float4 a = ((const float4*)shortcut)[gid];
  float4 v = ((const float4*)y)[wrow * 64 + c4];
  a.x += v.x; a.y += v.y; a.z += v.z; a.w += v.w;
  ((float4*)out)[gid] = a;
}

// ---------------------------------------------------------------------------
// Attention (swin windows & FPSI): one block = (sequence, head).
// q: per-lane global_load_async_to_lds_b32; k/v: TDM 2D tile loads.
// 128 threads = 4 waves; each wave owns rows i, i+4, ...
// ---------------------------------------------------------------------------
__global__ __launch_bounds__(128) void k_attn(
    const float* __restrict__ qkv, float* __restrict__ out, int n,
    const float* __restrict__ rpb, int masked) {
  __shared__ float sq[kWN * kDH], sk[kWN * kDH], sv[kWN * kDH];
  __shared__ float pbuf[4][kWN];
  const int seq = blockIdx.x >> 3, h = blockIdx.x & 7;
  const int tid = threadIdx.x;

  // low 32 bits of the generic shared pointer = LDS byte offset
  uint32_t lq = (uint32_t)(uintptr_t)sq;
  uint32_t lk = (uint32_t)(uintptr_t)sk;
  uint32_t lv = (uint32_t)(uintptr_t)sv;

  // k/v tiles via Tensor Data Mover (wave 0 issues; uniform descriptor)
  if (tid < 32) {
    const float* kbase = qkv + (size_t)seq * n * (3 * kEMB) + kEMB + h * kDH;
    const float* vbase = kbase + kEMB;
    tdm_load_2d(lk, kbase, n, kDH, 3 * kEMB);
    tdm_load_2d(lv, vbase, n, kDH, 3 * kEMB);
  }
  // q rows via per-lane async loads
  for (int idx = tid; idx < n * kDH; idx += 128) {
    int t = idx >> 5, d = idx & 31;
    const float* gp = qkv + ((size_t)seq * n + t) * (3 * kEMB) + h * kDH + d;
    uint64_t gq = (uint64_t)(uintptr_t)gp;
    asm volatile("global_load_async_to_lds_b32 %0, %1, off"
                 :: "v"(lq + (uint32_t)idx * 4u), "v"(gq) : "memory");
  }
  asm volatile("s_wait_asynccnt 0" ::: "memory");
  if (tid < 32) __builtin_amdgcn_s_wait_tensorcnt(0);
  __syncthreads();

  const int wid = tid >> 5, lane = tid & 31;
  int wh = 0, ww = 0, wd = 0;
  if (masked) {
    int wl = seq % 144;
    wh = wl / 24; ww = (wl >> 2) % 6; wd = wl & 3;
  }
  for (int i = wid; i < n; i += 4) {
    float sc[4];
#pragma unroll
    for (int t4 = 0; t4 < 4; ++t4) {
      int j = lane + t4 * 32;
      float s = -3.0e30f;
      if (j < n) {
        s = 0.0f;
#pragma unroll
        for (int d = 0; d < kDH; ++d) s += sq[i * kDH + d] * sk[j * kDH + d];
        s *= 0.17677669529663689f;  // dh^-0.5
        if (rpb) {
          int ii0 = i / 20, ii1 = (i >> 2) % 5, ii2 = i & 3;
          int jj0 = j / 20, jj1 = (j >> 2) % 5, jj2 = j & 3;
          int rel = ((ii0 - jj0 + 4) * 9 + (ii1 - jj1 + 4)) * 7 + (ii2 - jj2 + 3);
          s += rpb[rel * kHEADS + h];
        }
        if (masked) {
          int gi0 = wh * 5 + i / 20, gi1 = ww * 5 + (i >> 2) % 5, gi2 = wd * 4 + (i & 3);
          int gj0 = wh * 5 + j / 20, gj1 = ww * 5 + (j >> 2) % 5, gj2 = wd * 4 + (j & 3);
          int ri = (gi0 < 25 ? 0 : (gi0 < 28 ? 1 : 2)) * 9 +
                   (gi1 < 25 ? 0 : (gi1 < 28 ? 1 : 2)) * 3 +
                   (gi2 < 12 ? 0 : (gi2 < 14 ? 1 : 2));
          int rj = (gj0 < 25 ? 0 : (gj0 < 28 ? 1 : 2)) * 9 +
                   (gj1 < 25 ? 0 : (gj1 < 28 ? 1 : 2)) * 3 +
                   (gj2 < 12 ? 0 : (gj2 < 14 ? 1 : 2));
          if (ri != rj) s -= 100.0f;
        }
      }
      sc[t4] = s;
    }
    float mx = fmaxf(fmaxf(sc[0], sc[1]), fmaxf(sc[2], sc[3]));
#pragma unroll
    for (int o = 16; o > 0; o >>= 1) mx = fmaxf(mx, __shfl_xor(mx, o));
    float sum = 0.0f;
#pragma unroll
    for (int t4 = 0; t4 < 4; ++t4) {
      int j = lane + t4 * 32;
      float e = (j < n) ? __expf(sc[t4] - mx) : 0.0f;
      sc[t4] = e; sum += e;
    }
#pragma unroll
    for (int o = 16; o > 0; o >>= 1) sum += __shfl_xor(sum, o);
    float inv = 1.0f / sum;
#pragma unroll
    for (int t4 = 0; t4 < 4; ++t4) {
      int j = lane + t4 * 32;
      if (j < n) pbuf[wid][j] = sc[t4] * inv;
    }
    asm volatile("s_wait_dscnt 0" ::: "memory");  // same-wave LDS handoff
    float o = 0.0f;
    for (int j = 0; j < n; ++j) o += pbuf[wid][j] * sv[j * kDH + lane];
    out[((size_t)seq * n + i) * kEMB + h * kDH + lane] = o;
  }
}

// ---------------------------------------------------------------------------
// Patch embed im2col: rows = tokens, col = c*64 + i*16 + j*4 + k
// ---------------------------------------------------------------------------
__global__ void k_im2col(const float* __restrict__ x, float* __restrict__ A) {
  int gid = blockIdx.x * 256 + threadIdx.x;  // kTOK*64
  if (gid >= kTOK * 64) return;
  int m = gid >> 6, q = gid & 63;            // q = (c,i,j); k vectorized
  int c = q >> 4, i = (q >> 2) & 3, j = q & 3;
  int z = m & 15, y = (m >> 4) % 30, xp = (m / 480) % 30, b = m / 14400;
  size_t src = ((((size_t)(b * 4 + c) * 120 + (xp * 4 + i)) * 120) + (y * 4 + j)) * 64 +
               z * 4;
  ((float4*)A)[gid] = *(const float4*)(x + src);
}

// unpatch scatter: recon[b,c,h,w,d] = G[token, c*64+i*16+j*4+k] + unpatch_b[c]
__global__ void k_unpatch(const float* __restrict__ G, const float* __restrict__ ub,
                          float* __restrict__ out) {
  int gid = blockIdx.x * 256 + threadIdx.x;
  if (gid >= 1152000) return;
  int e = gid * 4;
  int d = e & 63, wq = (e >> 6) % 120, hq = (e / 7680) % 120, c = (e / 921600) % 4,
      b = e / 3686400;
  int xp = hq >> 2, i = hq & 3, y = wq >> 2, j = wq & 3, z = d >> 2;
  size_t tok = ((size_t)(b * 30 + xp) * 30 + y) * 16 + z;
  const float* gp = G + tok * 256 + (c * 64 + i * 16 + j * 4);
  float bb = ub[c];
  float4 v;
  v.x = gp[0] + bb; v.y = gp[1] + bb; v.z = gp[2] + bb; v.w = gp[3] + bb;
  ((float4*)out)[gid] = v;
}

// FPSI sequence gather (plane 1 = yz over x; plane 2 = xz over y)
__global__ void k_fpsi_gather(const float* __restrict__ feat, float* __restrict__ A,
                              int plane) {
  int gid = blockIdx.x * 256 + threadIdx.x;  // kTOK*64
  if (gid >= kTOK * 64) return;
  int c4 = gid & 63, row = gid >> 6;
  int n = row / 30, l = row % 30;
  int z = n & 15, u = (n >> 4) % 30, b = n / 480;
  int xp = (plane == 1) ? l : u;
  int y  = (plane == 1) ? u : l;
  size_t s = (((size_t)(b * 30 + xp) * 30 + y) * 16 + z) * 64 + c4;
  ((float4*)A)[gid] = ((const float4*)feat)[s];
}

// FPSI token assembly: t[n,0]=z_init+pos[0]; t[n,l+1]=pin[n,l]+pos[l+1]
__global__ void k_fpsi_build(const float* __restrict__ pin, const float* __restrict__ z0,
                             const float* __restrict__ pos, float* __restrict__ t,
                             int nseq, int L) {
  int gid = blockIdx.x * 256 + threadIdx.x;
  int total = nseq * (L + 1) * 64;
  if (gid >= total) return;
  int c4 = gid & 63, row = gid >> 6;
  int n = row / (L + 1), l = row % (L + 1);
  float4 pv = ((const float4*)pos)[l * 64 + c4];
  float4 v = (l == 0) ? ((const float4*)z0)[c4]
                      : ((const float4*)pin)[((size_t)n * L + (l - 1)) * 64 + c4];
  v.x += pv.x; v.y += pv.y; v.z += pv.z; v.w += pv.w;
  ((float4*)t)[gid] = v;
}

// gather class tokens t[:,0,:]
__global__ void k_fpsi_cls(const float* __restrict__ t, float* __restrict__ zr,
                           int nseq, int L1) {
  int gid = blockIdx.x * 256 + threadIdx.x;  // nseq*64
  if (gid >= nseq * 64) return;
  int c4 = gid & 63, n = gid >> 6;
  ((float4*)zr)[gid] = ((const float4*)t)[(size_t)n * L1 * 64 + c4];
}

// triplane fuse + dec_in projection (K=8, VALU): feat[tok,e]
__global__ void k_decfuse(const float* __restrict__ pxy, const float* __restrict__ pyz,
                          const float* __restrict__ pxz, const float* __restrict__ w8,
                          const float* __restrict__ b8, float* __restrict__ feat) {
  int gid = blockIdx.x * 256 + threadIdx.x;  // kTOK*256
  if (gid >= kTOK * kEMB) return;
  int e = gid & 255, tok = gid >> 8;
  int z = tok & 15, y = (tok >> 4) % 30, xp = (tok / 480) % 30, b = tok / 14400;
  int nxy = (b * 30 + xp) * 30 + y;
  int nyz = (b * 30 + y) * 16 + z;
  int nxz = (b * 30 + xp) * 16 + z;
  float acc = b8[e];
  const float* wr = w8 + e * 8;
#pragma unroll
  for (int o = 0; o < 8; ++o)
    acc += (pxy[nxy * 8 + o] + pyz[nyz * 8 + o] + pxz[nxz * 8 + o]) * wr[o];
  feat[gid] = acc;
}

// plane [nseq,8] -> d_out [b,8,D1,D2]
__global__ void k_plane_out(const float* __restrict__ pl, float* __restrict__ out,
                            int D1, int D2) {
  int gid = blockIdx.x * 256 + threadIdx.x;
  int total = kB * 8 * D1 * D2;
  if (gid >= total) return;
  int v = gid % D2, u = (gid / D2) % D1, o = (gid / (D1 * D2)) % 8,
      b = gid / (8 * D1 * D2);
  out[gid] = pl[((size_t)((b * D1 + u) * D2 + v)) * 8 + o];
}

// ---------------------------------------------------------------------------
// Host-side parameter bookkeeping (jax pytree sorted-key flatten order).
// ---------------------------------------------------------------------------
struct SwinP { const float *fc1_b,*fc1_w,*fc2_b,*fc2_w,*n1_b,*n1_g,*n2_b,*n2_g,
                           *proj_b,*proj_w,*qkv_b,*qkv_w,*rpb; };
struct FLay  { const float *l1_b,*l1_w,*l2_b,*l2_w,*n1_b,*n1_g,*n2_b,*n2_g,
                           *out_b,*out_w,*qkv_b,*qkv_w; };
struct FpsiP { FLay lay[2]; const float *pin_b,*pin_w,*pos,*pout_b,*pout_w,*z_init; };

extern "C" void kernel_launch(void* const* d_in, const int* in_sizes, int n_in,
                              void* d_out, int out_size, void* d_ws, size_t ws_size,
                              hipStream_t stream) {
  (void)out_size; (void)ws_size;
  int xi = 0;
  for (int i = 0; i < n_in; ++i)
    if (in_sizes[i] == kB * kCIN * kHh * kWw * kDd) { xi = i; break; }
  const float* x = (const float*)d_in[xi];

  int pi = 0;
  auto nxt = [&]() -> const float* {
    if (pi == xi) ++pi;
    return (const float*)d_in[pi++];
  };
  auto load_swin = [&](SwinP& s) {
    s.fc1_b = nxt(); s.fc1_w = nxt(); s.fc2_b = nxt(); s.fc2_w = nxt();
    s.n1_b = nxt(); s.n1_g = nxt(); s.n2_b = nxt(); s.n2_g = nxt();
    s.proj_b = nxt(); s.proj_w = nxt(); s.qkv_b = nxt(); s.qkv_w = nxt();
    s.rpb = nxt();
  };
  auto load_flay = [&](FLay& q) {
    q.l1_b = nxt(); q.l1_w = nxt(); q.l2_b = nxt(); q.l2_w = nxt();
    q.n1_b = nxt(); q.n1_g = nxt(); q.n2_b = nxt(); q.n2_g = nxt();
    q.out_b = nxt(); q.out_w = nxt(); q.qkv_b = nxt(); q.qkv_w = nxt();
  };
  auto load_fpsi = [&](FpsiP& f) {
    load_flay(f.lay[0]); load_flay(f.lay[1]);
    f.pin_b = nxt(); f.pin_w = nxt(); f.pos = nxt();
    f.pout_b = nxt(); f.pout_w = nxt(); f.z_init = nxt();
  };

  SwinP dec[4]; for (int i = 0; i < 4; ++i) load_swin(dec[i]);
  const float* dec_in_b = nxt();
  const float* dec_in_w = nxt();
  SwinP enc[4]; for (int i = 0; i < 4; ++i) load_swin(enc[i]);
  FpsiP fxy; load_fpsi(fxy);
  FpsiP fxz; load_fpsi(fxz);
  FpsiP fyz; load_fpsi(fyz);
  const float* patch_b = nxt();
  const float* patch_w = nxt();
  const float* unpatch_b = nxt();
  const float* unpatch_w = nxt();

  // -------- workspace layout (floats) --------
  float* wsf = (float*)d_ws;
  size_t off = 0;
  auto walloc = [&](size_t n) { float* p = wsf + off; off += n; return p; };
  const size_t MAXR = 30600;  // max rows across all GEMMs (1800*17)
  float* feat = walloc(MAXR * 256);
  float* tmp  = walloc(MAXR * 256);
  float* bufA = walloc(MAXR * 256);
  float* bufB = walloc(MAXR * 256);
  float* qkvb = walloc(MAXR * 768);
  float* hid  = walloc(MAXR * 1024);
  float* tseq = walloc(MAXR * 256);
  float* zrow = walloc(1800 * 256);
  float* pl_xy = walloc(1800 * 8);
  float* pl_yz = walloc(960 * 8);
  float* pl_xz = walloc(960 * 8);

  auto gemm = [&](const float* A, const float* Wt, const float* bias,
                  const float* res, float* C, int M, int N, int K, int wkn, int act) {
    dim3 g((N + 63) / 64, (M + 127) / 128);
    k_gemm<<<g, 256, 0, stream>>>(A, Wt, bias, res, C, M, N, K, wkn, act);
  };

  auto run_swin = [&](const SwinP& p, int masked) {
    k_layernorm<<<(kTOK + 7) / 8, 256, 0, stream>>>(feat, p.n1_g, p.n1_b, tmp, kTOK);
    k_winpart<<<kTOK * 64 / 256, 256, 0, stream>>>(tmp, bufA, masked);
    gemm(bufA, p.qkv_w, p.qkv_b, nullptr, qkvb, kTOK, 768, 256, 0, 0);
    k_attn<<<kNWIN * kHEADS, 128, 0, stream>>>(qkvb, bufB, kWN, p.rpb, masked);
    gemm(bufB, p.proj_w, p.proj_b, nullptr, bufA, kTOK, 256, 256, 0, 0);
    k_winrev<<<kTOK * 64 / 256, 256, 0, stream>>>(feat, bufA, tmp, masked);
    k_layernorm<<<(kTOK + 7) / 8, 256, 0, stream>>>(tmp, p.n2_g, p.n2_b, bufB, kTOK);
    gemm(bufB, p.fc1_w, p.fc1_b, nullptr, hid, kTOK, 1024, 256, 0, 1);
    gemm(hid, p.fc2_w, p.fc2_b, tmp, feat, kTOK, 256, 1024, 0, 0);
  };

  auto run_fpsi = [&](const FpsiP& p, int plane, int L, int nseq, float* pl) {
    const float* A = feat;  // xy plane: sequence gather is the identity layout
    if (plane != 0) {
      k_fpsi_gather<<<kTOK * 64 / 256, 256, 0, stream>>>(feat, bufA, plane);
      A = bufA;
    }
    gemm(A, p.pin_w, p.pin_b, nullptr, bufB, nseq * L, 256, 256, 0, 0);
    int rows = nseq * (L + 1);
    k_fpsi_build<<<(rows * 64 + 255) / 256, 256, 0, stream>>>(bufB, p.z_init, p.pos,
                                                              tseq, nseq, L);
    for (int li = 0; li < 2; ++li) {
      const FLay& q = p.lay[li];
      k_layernorm<<<(rows + 7) / 8, 256, 0, stream>>>(tseq, q.n1_g, q.n1_b, tmp, rows);
      gemm(tmp, q.qkv_w, q.qkv_b, nullptr, qkvb, rows, 768, 256, 0, 0);
      k_attn<<<nseq * kHEADS, 128, 0, stream>>>(qkvb, bufB, L + 1, nullptr, 0);
      gemm(bufB, q.out_w, q.out_b, tseq, tseq, rows, 256, 256, 0, 0);
      k_layernorm<<<(rows + 7) / 8, 256, 0, stream>>>(tseq, q.n2_g, q.n2_b, tmp, rows);
      gemm(tmp, q.l1_w, q.l1_b, nullptr, hid, rows, 1024, 256, 0, 1);
      gemm(hid, q.l2_w, q.l2_b, tseq, tseq, rows, 256, 1024, 0, 0);
    }
    k_fpsi_cls<<<(nseq * 64 + 255) / 256, 256, 0, stream>>>(tseq, zrow, nseq, L + 1);
    gemm(zrow, p.pout_w, p.pout_b, nullptr, pl, nseq, 8, 256, 0, 0);
  };

  // ---------------- forward ----------------
  k_im2col<<<kTOK * 64 / 256, 256, 0, stream>>>(x, bufA);
  gemm(bufA, patch_w, patch_b, nullptr, feat, kTOK, 256, 256, 0, 0);

  for (int i = 0; i < 4; ++i) run_swin(enc[i], i & 1);

  run_fpsi(fxy, 0, 16, 1800, pl_xy);
  run_fpsi(fyz, 1, 30, 960, pl_yz);
  run_fpsi(fxz, 2, 30, 960, pl_xz);

  k_decfuse<<<kTOK * 256 / 256, 256, 0, stream>>>(pl_xy, pl_yz, pl_xz, dec_in_w,
                                                  dec_in_b, feat);
  for (int i = 0; i < 4; ++i) run_swin(dec[i], i & 1);

  gemm(feat, unpatch_w, nullptr, nullptr, hid, kTOK, 256, 256, /*wkn=*/1, 0);

  float* outf = (float*)d_out;
  k_unpatch<<<(1152000 + 255) / 256, 256, 0, stream>>>(hid, unpatch_b, outf);
  k_plane_out<<<(14400 + 255) / 256, 256, 0, stream>>>(pl_xy, outf + 4608000, 30, 30);
  k_plane_out<<<(7680 + 255) / 256, 256, 0, stream>>>(pl_yz, outf + 4622400, 30, 16);
  k_plane_out<<<(7680 + 255) / 256, 256, 0, stream>>>(pl_xz, outf + 4630080, 30, 16);
}
